// VGAE3_2_72756745994352
// MI455X (gfx1250) — compile-verified, matrix operand order checked by reading, not run on hardware
//
#include <hip/hip_runtime.h>
#include <hip/hip_bf16.h>
#include <math.h>

// ---------------------------------------------------------------------------
// VGAE forward on MI455X (gfx1250, wave32).
//   * all operands bf16 (adj @128MB stays L2-resident across column blocks)
//   * universal bf16-WMMA GEMM, fp32 accumulate, double-buffered LDS
//   * A tiles (and B tiles in A*B^T mode) staged by the Tensor Data Mover
//     (tensor_load_to_lds + s_wait_tensorcnt), with TDM LDS padding matching
//     the 80B padded rows the WMMA fragment loads expect
//   * fused epilogues: relu / leaky+bias / sigmoid second output / bf16 copy
// ---------------------------------------------------------------------------

typedef __attribute__((ext_vector_type(16))) __bf16 v16bf;
typedef __attribute__((ext_vector_type(8)))  float  v8f;
typedef __attribute__((ext_vector_type(4)))  unsigned int u32x4;
typedef __attribute__((ext_vector_type(8)))  int    i32x8;
typedef __attribute__((ext_vector_type(4)))  int    i32x4;
typedef unsigned short u16;
typedef unsigned int   u32;

#define BM 128
#define BN 128
#define BK 32
#define LDS_ST 40   // u16 stride: 32 data + 8 pad -> 80B rows (16B aligned)

#if defined(__gfx1250__) && __has_builtin(__builtin_amdgcn_tensor_load_to_lds) && \
    __has_builtin(__builtin_amdgcn_s_wait_tensorcnt)
#define HAVE_TDM 1
#else
#define HAVE_TDM 0
#endif

__device__ __forceinline__ u16 f2bf(float f) {
    u32 u = __float_as_uint(f);
    u += 0x7FFFu + ((u >> 16) & 1u);        // round-to-nearest-even
    return (u16)(u >> 16);
}

union FragU { u32 u[8]; v16bf v; };

// 16x32 bf16 fragment from an LDS tile row-block.  Per ISA 16-bit A layout:
// lanes 0-15 -> K {0..7,16..23}, lanes 16-31 -> K {8..15,24..31}.
__device__ __forceinline__ void load_frag(const u16* rowp, int hf, FragU& f) {
    const u32* p = (const u32*)rowp;
    const int o = hf * 4;
    f.u[0] = p[0 + o];  f.u[1] = p[1 + o];
    f.u[2] = p[2 + o];  f.u[3] = p[3 + o];
    f.u[4] = p[8 + o];  f.u[5] = p[9 + o];
    f.u[6] = p[10 + o]; f.u[7] = p[11 + o];
}

#if HAVE_TDM
// TDM: load a 128-row x 32-col bf16 tile (row stride = rs elements) from
// gptr into LDS at lds_off, inserting 16B of padding after every 64B
// row (pad_interval=16 DWORDs -> code 3, pad_amount=4 DWORDs -> code 3) so
// the LDS image has the 80B row stride the fragment loads expect.
// Rows >= rows_rem are zero-filled by TDM's OOB handling.
__device__ __forceinline__ void tdm_tile_load(const u16* gptr, u32 lds_off,
                                              int rows_rem, int rs) {
    unsigned long long ga = (unsigned long long)(size_t)gptr;
    u32x4 g0;
    g0[0] = 1u;                                   // count=1 (valid descriptor)
    g0[1] = lds_off;                              // lds_addr
    g0[2] = (u32)(ga & 0xffffffffu);              // global_addr[31:0]
    g0[3] = (u32)((ga >> 32) & 0x01ffffffu)       // global_addr[56:32]
          | (2u << 30);                           // type = 2 ("image")
    i32x8 g1;
    const u32 dim0 = 32u;                         // tensor_dim0 (full cols)
    const u32 dim1 = (u32)rows_rem;               // tensor_dim1 -> OOB zeros
    g1[0] = (int)((1u << 16)                      // data_size = 2 bytes
                | (1u << 20)                      // pad_enable
                | (3u << 22)                      // pad_interval: 16 DWORDs
                | (3u << 25));                    // pad_amount: 4 DWORDs
    g1[1] = (int)((dim0 & 0xffffu) << 16);        // dim0[15:0]
    g1[2] = (int)(((dim0 >> 16) & 0xffffu) | ((dim1 & 0xffffu) << 16));
    g1[3] = (int)(((dim1 >> 16) & 0xffffu) | (32u << 16));  // tile_dim0=32
    g1[4] = (int)128u;                            // tile_dim1=128, tile_dim2=0
    g1[5] = (int)(u32)rs;                         // tensor_dim0_stride[31:0]
    g1[6] = 0;
    g1[7] = 0;
    const i32x4 zz4 = {0, 0, 0, 0};
    const i32x8 zz8 = {0, 0, 0, 0, 0, 0, 0, 0};
    // 6-arg form (amdgpu-toolchain / clang-23 signature)
    __builtin_amdgcn_tensor_load_to_lds(g0, g1, zz4, zz4, zz8, 0);
}
#endif

// ---------------------------------------------------------------------------
// f32 -> bf16 bulk convert (also used to concat the three mean-head weights)
__global__ __launch_bounds__(256)
void f32_to_bf16_kernel(const float* __restrict__ s, u16* __restrict__ d,
                        size_t n) {
    size_t i = ((size_t)blockIdx.x * blockDim.x + threadIdx.x) * 4;
    if (i >= n) return;
    float4 v = *(const float4*)(s + i);
    uint2 o;
    o.x = (u32)f2bf(v.x) | ((u32)f2bf(v.y) << 16);
    o.y = (u32)f2bf(v.z) | ((u32)f2bf(v.w) << 16);
    *(uint2*)(d + i) = o;
}

// ---------------------------------------------------------------------------
// act: 0 = identity, 1 = relu, 2 = leaky_relu(0.1) (bias added first if given)
// Cf/Cb/Csig (each optional): fp32, bf16, sigmoid(fp32) outputs.
// BNK (template): B stored [N,K] (A*B^T, TDM-staged); else [K,N] (VALU transpose).
template <int BNK>
__global__ __launch_bounds__(256)
void vgae_wmma_gemm_bf16(const u16* __restrict__ A, const u16* __restrict__ B,
                         const float* __restrict__ bias,
                         float* __restrict__ Cf, u16* __restrict__ Cb,
                         float* __restrict__ Csig,
                         int M, int N, int K, int act)
{
    __shared__ u16 As[2][BM][LDS_ST];   // [m][k]
    __shared__ u16 Bs[2][BN][LDS_ST];   // [n][k]  (n-major)

    const int t    = threadIdx.x;
    const int bm   = blockIdx.y;
    const int bn   = blockIdx.x;
    const int lane = t & 31;
    const int wave = t >> 5;
    const int wm   = wave & 3;      // 4 wave-rows of 32
    const int wn   = wave >> 2;     // 2 wave-cols of 64
    const int hf   = lane >> 4;
    const int lm   = lane & 15;

    v8f acc[2][4] = {};
    uint4 ar[2], br[2];
    const int nk = K >> 5;

    // ---- register staging helpers (fallback path & transposed-B path) ----
    auto stageA = [&](int k0) {
        #pragma unroll
        for (int i = 0; i < 2; ++i) {
            const int c   = t + i * 256;
            const int row = c >> 2;
            const int cb  = (c & 3) << 3;
            const int gr  = bm * BM + row;
            uint4 v = make_uint4(0u, 0u, 0u, 0u);
            if (gr < M) v = *(const uint4*)(A + (size_t)gr * K + k0 + cb);
            ar[i] = v;
        }
    };
    auto commitA = [&](int buf) {
        #pragma unroll
        for (int i = 0; i < 2; ++i) {
            const int c   = t + i * 256;
            const int row = c >> 2;
            const int cb  = (c & 3) << 3;
            *(uint4*)&As[buf][row][cb] = ar[i];
        }
    };
    auto stageB = [&](int k0) {
        #pragma unroll
        for (int i = 0; i < 2; ++i) {
            const int c = t + i * 256;
            uint4 v = make_uint4(0u, 0u, 0u, 0u);
            if (BNK) {                      // B[N,K]: coalesced along K
                const int row = c >> 2;
                const int cb  = (c & 3) << 3;
                const int gn  = bn * BN + row;
                if (gn < N) v = *(const uint4*)(B + (size_t)gn * K + k0 + cb);
            } else {                        // B[K,N]: coalesced along N
                const int k  = c >> 4;
                const int nb = (c & 15) << 3;
                const int gn = bn * BN + nb;
                if (gn < N) v = *(const uint4*)(B + (size_t)(k0 + k) * N + gn);
            }
            br[i] = v;
        }
    };
    auto commitB = [&](int buf) {
        #pragma unroll
        for (int i = 0; i < 2; ++i) {
            const int c = t + i * 256;
            if (BNK) {
                const int row = c >> 2;
                const int cb  = (c & 3) << 3;
                *(uint4*)&Bs[buf][row][cb] = br[i];
            } else {                        // transpose scatter into n-major
                const int k  = c >> 4;
                const int nb = (c & 15) << 3;
                u32 w[4] = { br[i].x, br[i].y, br[i].z, br[i].w };
                #pragma unroll
                for (int j = 0; j < 4; ++j) {
                    Bs[buf][nb + 2 * j + 0][k] = (u16)(w[j] & 0xffffu);
                    Bs[buf][nb + 2 * j + 1][k] = (u16)(w[j] >> 16);
                }
            }
        }
    };

    auto compute = [&](int buf) {
        FragU af[2], bfm[4];
        #pragma unroll
        for (int mi = 0; mi < 2; ++mi)
            load_frag(&As[buf][wm * 32 + mi * 16 + lm][0], hf, af[mi]);
        #pragma unroll
        for (int ni = 0; ni < 4; ++ni)
            load_frag(&Bs[buf][wn * 64 + ni * 16 + lm][0], hf, bfm[ni]);
        #pragma unroll
        for (int mi = 0; mi < 2; ++mi)
            #pragma unroll
            for (int ni = 0; ni < 4; ++ni)
                acc[mi][ni] = __builtin_amdgcn_wmma_f32_16x16x32_bf16(
                    false, af[mi].v, false, bfm[ni].v,
                    (short)0, acc[mi][ni], false, false);
    };

#if HAVE_TDM
    // ======== TDM-staged pipeline ========
    const u32 ldsA[2] = { (u32)(size_t)&As[0][0][0], (u32)(size_t)&As[1][0][0] };
    const u32 ldsB[2] = { (u32)(size_t)&Bs[0][0][0], (u32)(size_t)&Bs[1][0][0] };
    const int remM = M - bm * BM;
    const int remN = N - bn * BN;
    const u16* Abase = A + (size_t)bm * BM * K;
    const u16* Bbase = BNK ? (B + (size_t)bn * BN * K) : (const u16*)0;

    if (wave == 0) {
        tdm_tile_load(Abase, ldsA[0], remM, K);
        if (BNK) tdm_tile_load(Bbase, ldsB[0], remN, K);
    }
    if (!BNK) stageB(0);

    for (int it = 0; it < nk; ++it) {
        const int buf = it & 1;
        if (!BNK) commitB(buf);
        __syncthreads();                    // all waves done with buf^1
        if (wave == 0) {
            if (it + 1 < nk) {
                const int k0n = (it + 1) << 5;
                tdm_tile_load(Abase + k0n, ldsA[buf ^ 1], remM, K);
                if (BNK) tdm_tile_load(Bbase + k0n, ldsB[buf ^ 1], remN, K);
                __builtin_amdgcn_s_wait_tensorcnt(BNK ? 2 : 1);
            } else {
                __builtin_amdgcn_s_wait_tensorcnt(0);
            }
        }
        __syncthreads();                    // publish tile `it`
        if (!BNK && it + 1 < nk) stageB((it + 1) << 5);
        compute(buf);
    }
#else
    // ======== fallback: register-staged double buffering ========
    stageA(0); stageB(0);
    for (int it = 0; it < nk; ++it) {
        const int buf = it & 1;
        commitA(buf); commitB(buf);
        __syncthreads();
        if (it + 1 < nk) {
            const int k0n = (it + 1) << 5;
            stageA(k0n); stageB(k0n);
        }
        if (it + 2 < nk) {
            const int k0p = (it + 2) << 5;
            const int row = t >> 1;
            const int cb  = (t & 1) << 4;
            if (bm * BM + row < M)
                __builtin_prefetch(A + (size_t)(bm * BM + row) * K + k0p + cb, 0, 0);
            if (BNK) {
                if (bn * BN + row < N)
                    __builtin_prefetch(B + (size_t)(bn * BN + row) * K + k0p + cb, 0, 0);
            } else {
                const int k  = t >> 3;
                const int nb = (t & 7) << 4;
                if (bn * BN + nb < N)
                    __builtin_prefetch(B + (size_t)(k0p + k) * N + bn * BN + nb, 0, 0);
            }
        }
        compute(buf);
    }
#endif

    // ---- epilogue (C layout: VGPR r -> M = r + 8*hf, N = lane%16) ----
    #pragma unroll
    for (int mi = 0; mi < 2; ++mi) {
        #pragma unroll
        for (int ni = 0; ni < 4; ++ni) {
            const int rbase = bm * BM + wm * 32 + mi * 16 + hf * 8;
            const int cg    = bn * BN + wn * 64 + ni * 16 + lm;
            if (cg >= N) continue;
            const float bv = bias ? bias[cg] : 0.f;
            #pragma unroll
            for (int r = 0; r < 8; ++r) {
                const int rg = rbase + r;
                if (rg >= M) continue;
                float v = acc[mi][ni][r] + bv;
                float a;
                if      (act == 1) a = fmaxf(v, 0.f);
                else if (act == 2) a = (v > 0.f) ? v : 0.1f * v;
                else               a = v;
                const size_t idx = (size_t)rg * N + cg;
                if (Cf)   Cf[idx]   = a;
                if (Cb)   Cb[idx]   = f2bf(a);
                if (Csig) Csig[idx] = 1.f / (1.f + __expf(-a));
            }
        }
    }
}

// ---------------------------------------------------------------------------

extern "C" void kernel_launch(void* const* d_in, const int* in_sizes, int n_in,
                              void* d_out, int out_size, void* d_ws, size_t ws_size,
                              hipStream_t stream) {
    (void)in_sizes; (void)n_in; (void)out_size; (void)ws_size;

    const int N_ = 8192, IN = 1024, H1 = 512, H2 = 256, Z = 192;

    const float* adj = (const float*)d_in[0];
    const float* x   = (const float*)d_in[1];
    const float* W1  = (const float*)d_in[2];
    const float* W2  = (const float*)d_in[3];
    const float* Wm1 = (const float*)d_in[4];
    const float* Wm2 = (const float*)d_in[5];
    const float* Wm3 = (const float*)d_in[6];
    const float* Wt  = (const float*)d_in[7];
    const float* bt  = (const float*)d_in[8];

    float* out  = (float*)d_out;
    float* pred = out;                                   // [8192,8192]
    float* sig1 = out + (size_t)N_ * N_;                 // [8192,512]
    float* sig2 = sig1 + (size_t)N_ * H1;                // [8192,256]
    float* zout = sig2 + (size_t)N_ * H2;                // [8192,192]

    // bf16 workspace (bump allocation; all sizes multiples of 8 -> 16B aligned)
    u16* w = (u16*)d_ws;
    auto take = [&](size_t n) { u16* p = w; w += n; return p; };
    u16* adjb = take((size_t)N_ * N_);    // 128 MB -> L2 resident
    u16* xb   = take((size_t)N_ * IN);
    u16* W1b  = take((size_t)H1 * IN);
    u16* W2b  = take((size_t)H2 * H1);
    u16* Wmb  = take((size_t)Z  * H2);    // concat of 3 heads
    u16* Wtb  = take((size_t)Z  * Z);
    u16* B1b  = take((size_t)N_ * H1);
    u16* x1b  = take((size_t)N_ * H1);
    u16* B2b  = take((size_t)N_ * H2);
    u16* x2b  = take((size_t)N_ * H2);
    u16* Bmb  = take((size_t)N_ * Z);
    u16* zb   = take((size_t)N_ * Z);
    u16* Tb   = take((size_t)N_ * Z);

    const dim3 blk(256);
    auto cvt = [&](const float* s, u16* d, size_t n) {
        f32_to_bf16_kernel<<<dim3((unsigned)((n / 4 + 255) / 256)), blk, 0, stream>>>(s, d, n);
    };
    auto grid = [](int m, int n) { return dim3((n + BN - 1) / BN, (m + BM - 1) / BM); };

    // one-time bf16 conversion of all raw operands
    cvt(adj, adjb, (size_t)N_ * N_);
    cvt(x,   xb,   (size_t)N_ * IN);
    cvt(W1,  W1b,  (size_t)H1 * IN);
    cvt(W2,  W2b,  (size_t)H2 * H1);
    cvt(Wm1, Wmb,            (size_t)64 * H2);
    cvt(Wm2, Wmb +  64 * H2, (size_t)64 * H2);
    cvt(Wm3, Wmb + 128 * H2, (size_t)64 * H2);
    cvt(Wt,  Wtb,  (size_t)Z * Z);

    // 1) B1 = x * W1^T
    vgae_wmma_gemm_bf16<1><<<grid(N_, H1), blk, 0, stream>>>(
        xb, W1b, nullptr, nullptr, B1b, nullptr, N_, H1, IN, 0);
    // 2) x1 = relu(adj * B1);  sig1 = sigmoid(x1)
    vgae_wmma_gemm_bf16<0><<<grid(N_, H1), blk, 0, stream>>>(
        adjb, B1b, nullptr, nullptr, x1b, sig1, N_, H1, N_, 1);
    // 3) B2 = x1 * W2^T
    vgae_wmma_gemm_bf16<1><<<grid(N_, H2), blk, 0, stream>>>(
        x1b, W2b, nullptr, nullptr, B2b, nullptr, N_, H2, H1, 0);
    // 4) x2 = relu(adj * B2);  sig2 = sigmoid(x2)
    vgae_wmma_gemm_bf16<0><<<grid(N_, H2), blk, 0, stream>>>(
        adjb, B2b, nullptr, nullptr, x2b, sig2, N_, H2, N_, 1);
    // 5) Bm = x2 * Wm^T   (all three mean heads in one pass)
    vgae_wmma_gemm_bf16<1><<<grid(N_, Z), blk, 0, stream>>>(
        x2b, Wmb, nullptr, nullptr, Bmb, nullptr, N_, Z, H2, 0);
    // 6) z = adj * Bm  -> fp32 straight into d_out slice + bf16 for chaining
    vgae_wmma_gemm_bf16<0><<<grid(N_, Z), blk, 0, stream>>>(
        adjb, Bmb, nullptr, zout, zb, nullptr, N_, Z, N_, 0);
    // 7) T = leaky_relu(z * Wt^T + bt, 0.1)
    vgae_wmma_gemm_bf16<1><<<grid(N_, Z), blk, 0, stream>>>(
        zb, Wtb, bt, nullptr, Tb, nullptr, N_, Z, Z, 2);
    // 8) pred = z * T^T   (A,B both ~3MB: fully L2 resident; store-bound)
    vgae_wmma_gemm_bf16<1><<<grid(N_, N_), blk, 0, stream>>>(
        zb, Tb, nullptr, pred, nullptr, nullptr, N_, N_, Z, 0);
}
